// GRUD_cell_67877663146573
// MI455X (gfx1250) — matrix operand order repaired and checked by reading, not compiled, not run on hardware
//
#include <hip/hip_runtime.h>
#include <hip/hip_bf16.h>
#include <stdint.h>

// Problem constants (reference: B=64, I=128, H=256, T=1024)
#define BB   64
#define II   128
#define HH   256
#define TT   1024
#define RTOT (TT * BB)          // 65536 rows for the big GEMMs
#define LSTR 264                // padded LDS row stride, 528B -> conflict-free, 16B aligned

typedef __attribute__((ext_vector_type(16))) __bf16 v16bf;
typedef __attribute__((ext_vector_type(8)))  float  v8f;

union Frag {
    v16bf v;
    unsigned u[8];
    unsigned short us[16];
    uint4 q[2];
};

// Round two f32 to bf16 (RNE) and pack into one dword: lo in [15:0], hi in [31:16].
__device__ __forceinline__ unsigned pack_bf16(float lo, float hi) {
    unsigned ul = __float_as_uint(lo);
    ul = ul + 0x7FFFu + ((ul >> 16) & 1u);
    unsigned uh = __float_as_uint(hi);
    uh = uh + 0x7FFFu + ((uh >> 16) & 1u);
    return (ul >> 16) | (uh & 0xFFFF0000u);
}
__device__ __forceinline__ unsigned short f2bf(float f) {
    unsigned u = __float_as_uint(f);
    return (unsigned short)((u + 0x7FFFu + ((u >> 16) & 1u)) >> 16);
}
__device__ __forceinline__ float bf2f(unsigned short h) {
    return __uint_as_float(((unsigned)h) << 16);
}
__device__ __forceinline__ void pack8(const float4 a, const float4 b, unsigned* d) {
    d[0] = pack_bf16(a.x, a.y);
    d[1] = pack_bf16(a.z, a.w);
    d[2] = pack_bf16(b.x, b.y);
    d[3] = pack_bf16(b.z, b.w);
}
// Pack 16 f32 (two 8-element contiguous runs) into one bf16 fragment.
__device__ __forceinline__ void load_frag16(Frag& fr, const float* p0, const float* p1) {
    pack8(*(const float4*)p0, *(const float4*)(p0 + 4), fr.u);
    pack8(*(const float4*)p1, *(const float4*)(p1 + 4), fr.u + 4);
}
// 16 bf16 from LDS: runs at p and p+16 shorts (A-fragment K halves).
__device__ __forceinline__ void load_frag_lds(Frag& fr, const unsigned short* p) {
    fr.q[0] = *(const uint4*)p;
    fr.q[1] = *(const uint4*)(p + 16);
}
// 32B pre-packed fragment from global (fragment-ordered weight buffer).
__device__ __forceinline__ void load_frag_packed(Frag& fr, const uint4* p) {
    fr.q[0] = p[0];
    fr.q[1] = p[1];
}
__device__ __forceinline__ v8f bf_wmma(const Frag& a, const Frag& b, v8f c) {
    return __builtin_amdgcn_wmma_f32_16x16x32_bf16(false, a.v, false, b.v, (short)0, c,
                                                   false, false);
}
__device__ __forceinline__ float fast_sigmoid(float x) {
    return 1.f / (1.f + __expf(-x));
}
__device__ __forceinline__ float fast_tanh(float x) {
    return 1.f - 2.f / (__expf(x + x) + 1.f);
}

// ---------------------------------------------------------------------------
// Kernel 1: [B,I,T] -> [T,B,I] transpose of X, Mask, Delta (coalesced writes)
// ---------------------------------------------------------------------------
__global__ void k_transpose3(const float* __restrict__ X, const float* __restrict__ M,
                             const float* __restrict__ D,
                             float* __restrict__ Xt, float* __restrict__ Mt,
                             float* __restrict__ Dt) {
    size_t idx = (size_t)blockIdx.x * blockDim.x + threadIdx.x;
    if (idx >= (size_t)TT * BB * II) return;
    int i = (int)(idx % II);
    size_t r = idx / II;
    int b = (int)(r % BB);
    int t = (int)(r / BB);
    size_t src = ((size_t)b * II + i) * TT + t;
    Xt[idx] = X[src];
    Mt[idx] = M[src];
    Dt[idx] = D[src];
}

// ---------------------------------------------------------------------------
// Kernel 2: WMMA projection GEMM, compile-time specialized.
//   C[r, n] = act( A1[r,:]·W1[n,:] (+ A2[r,:]·W2[n,:]) + bias[n] )
//   EPI: 0=none, 1=+bias, 2=exp(-relu(x+bias)).  One 16x16 tile per wave.
// ---------------------------------------------------------------------------
template <int N, int EPI, bool FUSED>
__global__ __launch_bounds__(256) void k_proj_gemm(
    const float* __restrict__ A1, const float* __restrict__ A2,
    const float* __restrict__ W1, const float* __restrict__ W2,
    const float* __restrict__ bias, float* __restrict__ C) {
    const int lane  = threadIdx.x & 31;
    const int wave  = threadIdx.x >> 5;
    const int mtile = blockIdx.x;
    const int ntile = blockIdx.y * 8 + wave;

    const int mrow  = mtile * 16 + (lane & 15);
    const int ncol  = ntile * 16 + (lane & 15);
    const int kselA = (lane < 16) ? 0 : 8;    // A fragment K-half (ISA 7.12.2)
    const int khB   = (lane < 16) ? 0 : 8;    // B fragment K-half

    v8f acc = {};
#pragma unroll
    for (int kk = 0; kk < 128; kk += 32) {
        Frag a, b;
        const float* ap = A1 + (size_t)mrow * II + kk + kselA;
        load_frag16(a, ap, ap + 16);
        const float* wp = W1 + (size_t)ncol * II + kk + khB;
        load_frag16(b, wp, wp + 16);
        acc = bf_wmma(a, b, acc);
    }
    if (FUSED) {
#pragma unroll
        for (int kk = 0; kk < 128; kk += 32) {
            Frag a, b;
            const float* ap = A2 + (size_t)mrow * II + kk + kselA;
            load_frag16(a, ap, ap + 16);
            const float* wp = W2 + (size_t)ncol * II + kk + khB;
            load_frag16(b, wp, wp + 16);
            acc = bf_wmma(a, b, acc);
        }
    }
    float bv = 0.f;
    if (EPI >= 1) bv = bias[ncol];
    const int mbase = mtile * 16 + ((lane < 16) ? 0 : 8);
#pragma unroll
    for (int v = 0; v < 8; ++v) {
        float x = acc[v];
        if (EPI >= 1) x += bv;
        if (EPI == 2) x = __expf(-fmaxf(x, 0.f));
        C[(size_t)(mbase + v) * N + ncol] = x;
    }
}

// ---------------------------------------------------------------------------
// Kernel 3: per-(b,i) time scan producing x2 (GRU-D imputation)
// ---------------------------------------------------------------------------
__global__ void k_scan_impute(const float* __restrict__ Xt, const float* __restrict__ Mt,
                              const float* __restrict__ GX, const float* __restrict__ xmean_p,
                              float* __restrict__ X2) {
    int tid = blockIdx.x * blockDim.x + threadIdx.x;   // over B*I
    if (tid >= BB * II) return;
    const float xmean = *xmean_p;
    float xl = 0.f;
    for (int t = 0; t < TT; ++t) {
        size_t o = (size_t)t * BB * II + tid;
        float x  = Xt[o];
        float m  = Mt[o];
        float gx = GX[o];
        xl = (m > 0.f) ? x : xl;                        // update BEFORE imputation
        float x1 = m * x  + (1.f - m) * (gx * x  + (1.f - gx) * xmean);
        float x2 = m * x1 + (1.f - m) * (gx * xl + (1.f - gx) * xmean);
        X2[o] = x2;
    }
}

// ---------------------------------------------------------------------------
// Kernel 4: pack W_hz/W_hr/W_hh into fragment-ordered bf16 buffer.
//   Slot s = ((g*16 + nt)*8 + kk)*32 + lane  ->  32 bytes (16 bf16) per lane.
// ---------------------------------------------------------------------------
__global__ void k_pack_weights(const float* __restrict__ Whz, const float* __restrict__ Whr,
                               const float* __restrict__ Whh, unsigned* __restrict__ WB) {
    int s = blockIdx.x * blockDim.x + threadIdx.x;     // 3*16*8*32 = 12288 slots
    if (s >= 3 * 16 * 8 * 32) return;
    int lane = s & 31;
    int kk   = (s >> 5) & 7;
    int nt   = (s >> 8) & 15;
    int g    = s >> 12;
    const float* W = (g == 0) ? Whz : (g == 1) ? Whr : Whh;
    int ncol = nt * 16 + (lane & 15);
    int khB  = (lane < 16) ? 0 : 8;
    const float* p = W + (size_t)ncol * HH + kk * 32 + khB;
    Frag f;
    load_frag16(f, p, p + 16);
#pragma unroll
    for (int j = 0; j < 8; ++j) WB[(size_t)s * 8 + j] = f.u[j];
}

// ---------------------------------------------------------------------------
// Kernel 5: sequential recurrence.
//   4 blocks (16-batch tiles) x 256 threads (8 waves); each wave owns 2
//   N-tiles (wave, wave+8) sharing one set of A-fragments. Weight B-fragments
//   stream from the L2-resident pre-packed buffer; LICM promotes them into
//   high VGPRs (s_set_vgpr_msb) without forced spills. 48 WMMA/wave/step.
//   hd-phase is fully vectorized: b128 loads + b128 LDS stores per row chunk.
// ---------------------------------------------------------------------------
__global__ __launch_bounds__(256) void k_recurrence(
    const float* __restrict__ GH, const float* __restrict__ AZ,
    const float* __restrict__ AR, const float* __restrict__ AH,
    const uint4* __restrict__ WB, float* __restrict__ out /* [B,T,H] */) {
    __shared__ __attribute__((aligned(16))) float          hbuf[16 * LSTR];
    __shared__ __attribute__((aligned(16))) unsigned short hdbuf[16 * LSTR];
    __shared__ __attribute__((aligned(16))) unsigned short rhdbuf[16 * LSTR];

    const int tid   = threadIdx.x;
    const int lane  = tid & 31;
    const int wave  = tid >> 5;                 // 0..7
    const int btile = blockIdx.x;               // 0..3 -> 16-batch tile
    const int ksel  = (lane < 16) ? 0 : 8;
    const int moff  = (lane < 16) ? 0 : 8;

    // This wave's two N-tiles and per-tile output columns.
    const int nt0   = wave;
    const int nt1   = wave + 8;
    const int ncol0 = nt0 * 16 + (lane & 15);
    const int ncol1 = nt1 * 16 + (lane & 15);

    // hd-phase mapping: one 16-column row chunk per thread.
    const int hm  = tid >> 4;                   // row 0..15
    const int hn0 = (tid & 15) * 16;            // column base

    // Fragment-slot bases: slot = ((g*16 + nt)*8 + kk)*32 + lane; 2 uint4 per slot.
    #define WSLOT(g, nt, kk) (WB + ((size_t)((((g)*16 + (nt))*8 + (kk))*32 + lane)) * 2)

    for (int j = tid; j < 16 * LSTR; j += 256) hbuf[j] = 0.f;   // h0 = 0

    // Prefetch next timestep's gate rows: 64 threads per tensor, 2 lines each.
    const float* pf_base = (tid < 64) ? GH : (tid < 128) ? AZ : (tid < 192) ? AR : AH;
    const int    pf_off  = (tid & 63) * 32;

    for (int t = 0; t < TT; ++t) {
        __syncthreads();                                        // h writes visible
        if (t + 1 < TT) {
            size_t nbase = ((size_t)(t + 1) * BB + btile * 16) * HH;
            __builtin_prefetch(pf_base + nbase + pf_off, 0, 0);        // global_prefetch_b8
            __builtin_prefetch(pf_base + nbase + pf_off + 2048, 0, 0);
        }
        // hd = gamma_h * h_prev -> bf16 LDS (vectorized: 4x b128 ld + 2x b128 st)
        {
            const float4* ghp = (const float4*)(GH + ((size_t)t * BB + btile * 16 + hm) * HH + hn0);
            const float4* hp  = (const float4*)(hbuf + hm * LSTR + hn0);
            unsigned pk[8];
#pragma unroll
            for (int q = 0; q < 4; ++q) {
                float4 g = ghp[q];
                float4 h = hp[q];
                pk[2 * q]     = pack_bf16(g.x * h.x, g.y * h.y);
                pk[2 * q + 1] = pack_bf16(g.z * h.z, g.w * h.w);
            }
            uint4* dst = (uint4*)(hdbuf + hm * LSTR + hn0);
            dst[0] = make_uint4(pk[0], pk[1], pk[2], pk[3]);
            dst[1] = make_uint4(pk[4], pk[5], pk[6], pk[7]);
        }
        __syncthreads();

        Frag a[8];
#pragma unroll
        for (int kk = 0; kk < 8; ++kk)
            load_frag_lds(a[kk], hdbuf + (lane & 15) * LSTR + kk * 32 + ksel);

        const size_t gbase = ((size_t)t * BB + btile * 16) * HH;
        float zv[2][8], hdv[2][8];
#pragma unroll
        for (int nt = 0; nt < 2; ++nt) {
            const int ntile = nt ? nt1 : nt0;
            const int ncol  = nt ? ncol1 : ncol0;
            v8f zacc = {}, racc = {};
#pragma unroll
            for (int kk = 0; kk < 8; ++kk) {
                Frag w;
                load_frag_packed(w, WSLOT(0, ntile, kk));
                zacc = bf_wmma(a[kk], w, zacc);
            }
#pragma unroll
            for (int kk = 0; kk < 8; ++kk) {
                Frag w;
                load_frag_packed(w, WSLOT(1, ntile, kk));
                racc = bf_wmma(a[kk], w, racc);
            }
#pragma unroll
            for (int v = 0; v < 8; ++v) {
                int m = v + moff;
                float az = AZ[gbase + (size_t)m * HH + ncol];
                float ar = AR[gbase + (size_t)m * HH + ncol];
                float z  = fast_sigmoid(zacc[v] + az);
                float r  = fast_sigmoid(racc[v] + ar);
                float hd = bf2f(hdbuf[m * LSTR + ncol]);
                zv[nt][v] = z; hdv[nt][v] = hd;
                rhdbuf[m * LSTR + ncol] = f2bf(r * hd);
            }
        }
        __syncthreads();

        Frag ra[8];
#pragma unroll
        for (int kk = 0; kk < 8; ++kk)
            load_frag_lds(ra[kk], rhdbuf + (lane & 15) * LSTR + kk * 32 + ksel);
#pragma unroll
        for (int nt = 0; nt < 2; ++nt) {
            const int ntile = nt ? nt1 : nt0;
            const int ncol  = nt ? ncol1 : ncol0;
            v8f hacc = {};
#pragma unroll
            for (int kk = 0; kk < 8; ++kk) {
                Frag w;
                load_frag_packed(w, WSLOT(2, ntile, kk));
                hacc = bf_wmma(ra[kk], w, hacc);
            }
#pragma unroll
            for (int v = 0; v < 8; ++v) {
                int m = v + moff;
                float ah = AH[gbase + (size_t)m * HH + ncol];
                float ht = fast_tanh(hacc[v] + ah);
                float hn = (1.f - zv[nt][v]) * hdv[nt][v] + zv[nt][v] * ht;
                hbuf[m * LSTR + ncol] = hn;
                out[((size_t)(btile * 16 + m) * TT + t) * HH + ncol] = hn;
            }
        }
    }
    #undef WSLOT
}

// ---------------------------------------------------------------------------
extern "C" void kernel_launch(void* const* d_in, const int* in_sizes, int n_in,
                              void* d_out, int out_size, void* d_ws, size_t ws_size,
                              hipStream_t stream) {
    const float* X      = (const float*)d_in[0];
    const float* Mask   = (const float*)d_in[1];
    const float* Delta  = (const float*)d_in[2];
    const float* xmean  = (const float*)d_in[3];
    const float* W_dg_x = (const float*)d_in[4];
    const float* b_dg_x = (const float*)d_in[5];
    const float* W_dg_h = (const float*)d_in[6];
    const float* b_dg_h = (const float*)d_in[7];
    const float* W_xz   = (const float*)d_in[8];
    const float* W_hz   = (const float*)d_in[9];
    const float* W_mz   = (const float*)d_in[10];
    const float* b_mz   = (const float*)d_in[11];
    const float* W_xr   = (const float*)d_in[12];
    const float* W_hr   = (const float*)d_in[13];
    const float* W_mr   = (const float*)d_in[14];
    const float* W_xh   = (const float*)d_in[15];
    const float* W_hh   = (const float*)d_in[16];
    const float* W_mh   = (const float*)d_in[17];
    const float* b_mh   = (const float*)d_in[18];
    float* out = (float*)d_out;

    // Workspace partition (f32 unless noted)
    const size_t NI = (size_t)RTOT * II;   // 8.39M
    const size_t NH = (size_t)RTOT * HH;   // 16.78M
    float* Xt = (float*)d_ws;
    float* Mt = Xt + NI;
    float* Dt = Mt + NI;
    float* GX = Dt + NI;
    float* X2 = GX + NI;
    float* GH = X2 + NI;
    float* AZ = GH + NH;
    float* AR = AZ + NH;
    float* AH = AR + NH;
    unsigned* WB = (unsigned*)(AH + NH);   // 384KB packed bf16 weight fragments

    // 1) transpose inputs to [T,B,I]
    {
        size_t n = (size_t)TT * BB * II;
        k_transpose3<<<dim3((unsigned)((n + 255) / 256)), dim3(256), 0, stream>>>(
            X, Mask, Delta, Xt, Mt, Dt);
    }
    // 2) gamma_x = exp(-relu(Dt @ W_dg_x^T + b))   [RTOT x 128]
    k_proj_gemm<II, 2, false><<<dim3(RTOT / 16, 1), dim3(256), 0, stream>>>(
        Dt, nullptr, W_dg_x, nullptr, b_dg_x, GX);
    // 3) imputation scan -> X2
    k_scan_impute<<<dim3((BB * II) / 256), dim3(256), 0, stream>>>(Xt, Mt, GX, xmean, X2);
    // 4) gamma_h = exp(-relu(Dt @ W_dg_h^T + b))   [RTOT x 256]
    k_proj_gemm<HH, 2, false><<<dim3(RTOT / 16, 2), dim3(256), 0, stream>>>(
        Dt, nullptr, W_dg_h, nullptr, b_dg_h, GH);
    // 5) AZ = X2@W_xz^T + Mask@W_mz^T + b_mz
    k_proj_gemm<HH, 1, true><<<dim3(RTOT / 16, 2), dim3(256), 0, stream>>>(
        X2, Mt, W_xz, W_mz, b_mz, AZ);
    // 6) AR = X2@W_xr^T + Mask@W_mr^T
    k_proj_gemm<HH, 0, true><<<dim3(RTOT / 16, 2), dim3(256), 0, stream>>>(
        X2, Mt, W_xr, W_mr, nullptr, AR);
    // 7) AH = X2@W_xh^T + Mask@W_mh^T + b_mh
    k_proj_gemm<HH, 1, true><<<dim3(RTOT / 16, 2), dim3(256), 0, stream>>>(
        X2, Mt, W_xh, W_mh, b_mh, AH);
    // 8) pack recurrent weights into WMMA fragment order (bf16)
    k_pack_weights<<<dim3(48), dim3(256), 0, stream>>>(W_hz, W_hr, W_hh, WB);
    // 9) sequential recurrence -> out [B,T,H]
    k_recurrence<<<dim3(BB / 16), dim3(256), 0, stream>>>(
        GH, AZ, AR, AH, (const uint4*)WB, out);
    (void)in_sizes; (void)n_in; (void)out_size; (void)ws_size;
}
